// RNN_28037546508953
// MI455X (gfx1250) — matrix operand order repaired and checked by reading, not compiled, not run on hardware
//
#include <hip/hip_runtime.h>
#include <hip/hip_bf16.h>
#include <math.h>

#define B_ 256
#define S_ 512
#define E_ 256
#define H_ 512
#define C_ 128
#define BS_ (B_ * S_)          // 131072 flattened (b,t) rows
#define G3H_ (3 * H_)          // 1536

typedef __attribute__((ext_vector_type(16))) __bf16 v16bf;
typedef __attribute__((ext_vector_type(8)))  __bf16 v8bf;
typedef __attribute__((ext_vector_type(8)))  float  v8f;

// ---------------------------------------------------------------------------
// Fragment loaders (wave32, v_wmma_f32_16x16x32_bf16 operand layouts)
//
// A (16x32, MxK, 16-bit): lane L holds row M=L&15.
//   halves 0..7  = K = k0 + (L>>4)*8 + [0..7]
//   halves 8..15 = K = k0 + 16 + (L>>4)*8 + [0..7]
// B (32x16, KxN, 16-bit): lane L holds column N=L&15.
//   halves 0..7 = K = k0 + (L>>4)*8 + [0..7], halves 8..15 = K = k0 + 16 + (L>>4)*8 + [0..7]
// C/D (16x16 f32): lane L holds column N=L&15, VGPR v holds row M=(L>>4)*8+v.
// ---------------------------------------------------------------------------

__device__ __forceinline__ v16bf cat8(v8bf lo, v8bf hi) {
  return __builtin_shufflevector(lo, hi, 0,1,2,3,4,5,6,7,8,9,10,11,12,13,14,15);
}

__device__ __forceinline__ v16bf load_a_f32(const float* __restrict__ base,
                                            int row, long ld, int k0, int lane) {
  const float* p = base + (long)row * ld + k0 + ((lane >> 4) << 3);
  v8f x0 = *(const v8f*)p;
  v8f x1 = *(const v8f*)(p + 16);
  v16bf r;
#pragma unroll
  for (int i = 0; i < 8; ++i) {
    r[i]     = (__bf16)x0[i];
    r[i + 8] = (__bf16)x1[i];
  }
  return r;
}

__device__ __forceinline__ v16bf load_a_bf16(const __bf16* __restrict__ base,
                                             int row, long ld, int k0, int lane) {
  const __bf16* p = base + (long)row * ld + k0 + ((lane >> 4) << 3);
  v8bf lo = *(const v8bf*)p;
  v8bf hi = *(const v8bf*)(p + 16);
  return cat8(lo, hi);
}

__device__ __forceinline__ v16bf load_b_bf16(const __bf16* __restrict__ base,
                                             int wrow, long ld, int k0, int lane) {
  const __bf16* p = base + (long)wrow * ld + k0 + ((lane >> 4) << 3);
  v8bf lo = *(const v8bf*)p;
  v8bf hi = *(const v8bf*)(p + 16);
  return cat8(lo, hi);
}

__device__ __forceinline__ v16bf load_b_f32(const float* __restrict__ base,
                                            int wrow, long ld, int k0, int lane) {
  const float* p = base + (long)wrow * ld + k0 + ((lane >> 4) << 3);
  v8f x0 = *(const v8f*)p;
  v8f x1 = *(const v8f*)(p + 16);
  v16bf r;
#pragma unroll
  for (int i = 0; i < 8; ++i) {
    r[i]     = (__bf16)x0[i];
    r[i + 8] = (__bf16)x1[i];
  }
  return r;
}

__device__ __forceinline__ v8f wmma_bf16(v16bf a, v16bf b, v8f c) {
  return __builtin_amdgcn_wmma_f32_16x16x32_bf16(false, a, false, b,
                                                 (short)0, c, false, false);
}

__device__ __forceinline__ v8f splat8(float s) {
  v8f r;
#pragma unroll
  for (int i = 0; i < 8; ++i) r[i] = s;
  return r;
}

__device__ __forceinline__ float sigmoidf_(float x) {
  return 1.0f / (1.0f + __expf(-x));
}

// ---------------------------------------------------------------------------
// One-time fp32 -> bf16 conversion of recurrent weights + h0.
// ---------------------------------------------------------------------------
__global__ void convert_inputs(const float* __restrict__ Wih,
                               const float* __restrict__ Whh,
                               const float* __restrict__ h0,
                               __bf16* __restrict__ Wih_bf,
                               __bf16* __restrict__ Whh_bf,
                               __bf16* __restrict__ h0_bf) {
  int i = blockIdx.x * blockDim.x + threadIdx.x;
  if (i < G3H_ * E_) Wih_bf[i] = (__bf16)Wih[i];
  if (i < G3H_ * H_) Whh_bf[i] = (__bf16)Whh[i];
  if (i < B_ * H_)   h0_bf[i]  = (__bf16)h0[i];
}

// fp32 -> bf16 conversion of the whole input stream X [B*S, E].
__global__ void convert_x(const float* __restrict__ x, __bf16* __restrict__ xbf) {
  int i = blockIdx.x * blockDim.x + threadIdx.x;
  if (i < BS_ * E_) xbf[i] = (__bf16)x[i];
}

// ---------------------------------------------------------------------------
// GI = X @ W_ih^T + b_ih, fully parallel over all (b,t) rows (off the scan's
// critical path). A rows of the flattened [B*S, E] X map 1:1 onto memory.
// grid = (BS/32, 3H/16), block = 32, 2 M-tiles per wave share B fragments.
// ---------------------------------------------------------------------------
__global__ void __launch_bounds__(32)
gi_gemm(const __bf16* __restrict__ Xbf,
        const __bf16* __restrict__ Wih_bf,
        const float* __restrict__ b_ih,
        float* __restrict__ GI) {
  const int lane = threadIdx.x;
  const int m0 = blockIdx.x * 32;
  const int j0 = blockIdx.y * 16;
  const int jc = j0 + (lane & 15);

  v8f acc0 = splat8(b_ih[jc]);
  v8f acc1 = acc0;
  const int arow0 = m0 + (lane & 15);
  const int arow1 = arow0 + 16;

#pragma unroll 2
  for (int k0 = 0; k0 < E_; k0 += 32) {
    v16bf a0 = load_a_bf16(Xbf, arow0, E_, k0, lane);
    v16bf a1 = load_a_bf16(Xbf, arow1, E_, k0, lane);
    v16bf b  = load_b_bf16(Wih_bf, jc, E_, k0, lane);
    acc0 = wmma_bf16(a0, b, acc0);
    acc1 = wmma_bf16(a1, b, acc1);
  }

#pragma unroll
  for (int mt = 0; mt < 2; ++mt) {
    const v8f& ac = mt ? acc1 : acc0;
    const int mbase = m0 + 16 * mt + ((lane >> 4) << 3);
#pragma unroll
    for (int v = 0; v < 8; ++v)
      GI[(long)(mbase + v) * G3H_ + jc] = ac[v];
  }
}

// ---------------------------------------------------------------------------
// Recurrent step, GI-precomputed tier: only gh = h @ W_hh^T stays serialized.
// All GEMM operands bf16; gate fusion reads GI rows (f32) + h_old (f32).
// grid = (H/16, B/32), block = 32.
// ---------------------------------------------------------------------------
__global__ void __launch_bounds__(32)
gru_step_gh(const float* __restrict__ GI, int t,
            const float* __restrict__ hprev_f,
            const __bf16* __restrict__ hprev_b,
            float* __restrict__ hnext_f,
            __bf16* __restrict__ hnext_b,
            const __bf16* __restrict__ Whh_bf,
            const float* __restrict__ b_hh) {
  const int lane = threadIdx.x;
  const int j0 = blockIdx.x * 16;
  const int m0 = blockIdx.y * 32;
  const int jc = j0 + (lane & 15);

  v8f acc_r0  = splat8(b_hh[jc]),          acc_r1  = acc_r0;
  v8f acc_z0  = splat8(b_hh[H_ + jc]),     acc_z1  = acc_z0;
  v8f acc_hn0 = splat8(b_hh[2 * H_ + jc]), acc_hn1 = acc_hn0;

  const int arow0 = m0 + (lane & 15);
  const int arow1 = arow0 + 16;

#pragma unroll 2
  for (int k0 = 0; k0 < H_; k0 += 32) {
    v16bf a0 = load_a_bf16(hprev_b, arow0, H_, k0, lane);
    v16bf a1 = load_a_bf16(hprev_b, arow1, H_, k0, lane);
    v16bf br = load_b_bf16(Whh_bf, jc,          H_, k0, lane);
    v16bf bz = load_b_bf16(Whh_bf, H_ + jc,     H_, k0, lane);
    v16bf bn = load_b_bf16(Whh_bf, 2 * H_ + jc, H_, k0, lane);
    acc_r0  = wmma_bf16(a0, br, acc_r0);
    acc_r1  = wmma_bf16(a1, br, acc_r1);
    acc_z0  = wmma_bf16(a0, bz, acc_z0);
    acc_z1  = wmma_bf16(a1, bz, acc_z1);
    acc_hn0 = wmma_bf16(a0, bn, acc_hn0);
    acc_hn1 = wmma_bf16(a1, bn, acc_hn1);
  }

#pragma unroll
  for (int mt = 0; mt < 2; ++mt) {
    const v8f& ar = mt ? acc_r1 : acc_r0;
    const v8f& az = mt ? acc_z1 : acc_z0;
    const v8f& ah = mt ? acc_hn1 : acc_hn0;
    const int mbase = m0 + 16 * mt + ((lane >> 4) << 3);
#pragma unroll
    for (int v = 0; v < 8; ++v) {
      const int m = mbase + v;
      const long gbase = ((long)m * S_ + t) * G3H_;
      float gr = GI[gbase + jc];
      float gz = GI[gbase + H_ + jc];
      float gn = GI[gbase + 2 * H_ + jc];
      const long idx = (long)m * H_ + jc;
      float h_old = hprev_f[idx];
      float r = sigmoidf_(gr + ar[v]);
      float z = sigmoidf_(gz + az[v]);
      float n = tanhf(gn + r * ah[v]);
      float h = (1.0f - z) * n + z * h_old;
      hnext_f[idx] = h;
      hnext_b[idx] = (__bf16)h;
    }
  }
}

// ---------------------------------------------------------------------------
// Fused step (fallback tiers): gi + gh in one kernel. X_IS_BF16 selects
// whether the x operand comes pre-converted (tier 2) or as f32 (tier 3).
// grid = (H/16, B/32), block = 32.
// ---------------------------------------------------------------------------
template <bool X_IS_BF16>
__global__ void __launch_bounds__(32)
gru_step_fused(const void* __restrict__ x_t_, long x_row_stride,
               const float* __restrict__ hprev_f,
               const __bf16* __restrict__ hprev_b,
               float* __restrict__ hnext_f,
               __bf16* __restrict__ hnext_b,
               const __bf16* __restrict__ Wih_bf,
               const __bf16* __restrict__ Whh_bf,
               const float* __restrict__ b_ih,
               const float* __restrict__ b_hh) {
  const int lane = threadIdx.x;
  const int j0 = blockIdx.x * 16;
  const int m0 = blockIdx.y * 32;
  const int jc = j0 + (lane & 15);

  const float bias_r  = b_ih[jc]          + b_hh[jc];
  const float bias_z  = b_ih[H_ + jc]     + b_hh[H_ + jc];
  v8f acc_r0  = splat8(bias_r),               acc_r1  = acc_r0;
  v8f acc_z0  = splat8(bias_z),               acc_z1  = acc_z0;
  v8f acc_in0 = splat8(b_ih[2 * H_ + jc]),    acc_in1 = acc_in0;
  v8f acc_hn0 = splat8(b_hh[2 * H_ + jc]),    acc_hn1 = acc_hn0;

  const int arow0 = m0 + (lane & 15);
  const int arow1 = arow0 + 16;

#pragma unroll 2
  for (int k0 = 0; k0 < E_; k0 += 32) {
    v16bf a0, a1;
    if (X_IS_BF16) {
      a0 = load_a_bf16((const __bf16*)x_t_, arow0, x_row_stride, k0, lane);
      a1 = load_a_bf16((const __bf16*)x_t_, arow1, x_row_stride, k0, lane);
    } else {
      a0 = load_a_f32((const float*)x_t_, arow0, x_row_stride, k0, lane);
      a1 = load_a_f32((const float*)x_t_, arow1, x_row_stride, k0, lane);
    }
    v16bf br = load_b_bf16(Wih_bf, jc,          E_, k0, lane);
    v16bf bz = load_b_bf16(Wih_bf, H_ + jc,     E_, k0, lane);
    v16bf bn = load_b_bf16(Wih_bf, 2 * H_ + jc, E_, k0, lane);
    acc_r0  = wmma_bf16(a0, br, acc_r0);
    acc_r1  = wmma_bf16(a1, br, acc_r1);
    acc_z0  = wmma_bf16(a0, bz, acc_z0);
    acc_z1  = wmma_bf16(a1, bz, acc_z1);
    acc_in0 = wmma_bf16(a0, bn, acc_in0);
    acc_in1 = wmma_bf16(a1, bn, acc_in1);
  }

#pragma unroll 2
  for (int k0 = 0; k0 < H_; k0 += 32) {
    v16bf a0 = load_a_bf16(hprev_b, arow0, H_, k0, lane);
    v16bf a1 = load_a_bf16(hprev_b, arow1, H_, k0, lane);
    v16bf br = load_b_bf16(Whh_bf, jc,          H_, k0, lane);
    v16bf bz = load_b_bf16(Whh_bf, H_ + jc,     H_, k0, lane);
    v16bf bn = load_b_bf16(Whh_bf, 2 * H_ + jc, H_, k0, lane);
    acc_r0  = wmma_bf16(a0, br, acc_r0);
    acc_r1  = wmma_bf16(a1, br, acc_r1);
    acc_z0  = wmma_bf16(a0, bz, acc_z0);
    acc_z1  = wmma_bf16(a1, bz, acc_z1);
    acc_hn0 = wmma_bf16(a0, bn, acc_hn0);
    acc_hn1 = wmma_bf16(a1, bn, acc_hn1);
  }

#pragma unroll
  for (int mt = 0; mt < 2; ++mt) {
    const v8f& ar = mt ? acc_r1 : acc_r0;
    const v8f& az = mt ? acc_z1 : acc_z0;
    const v8f& ai = mt ? acc_in1 : acc_in0;
    const v8f& ah = mt ? acc_hn1 : acc_hn0;
    const int mbase = m0 + 16 * mt + ((lane >> 4) << 3);
#pragma unroll
    for (int v = 0; v < 8; ++v) {
      const long idx = (long)(mbase + v) * H_ + jc;
      float h_old = hprev_f[idx];
      float r = sigmoidf_(ar[v]);
      float z = sigmoidf_(az[v]);
      float n = tanhf(ai[v] + r * ah[v]);
      float h = (1.0f - z) * n + z * h_old;
      hnext_f[idx] = h;
      hnext_b[idx] = (__bf16)h;
    }
  }
}

// ---------------------------------------------------------------------------
// logits = h_last @ W_out^T + b_out.  grid = (C/16, B/16), block = 32.
// ---------------------------------------------------------------------------
__global__ void __launch_bounds__(32)
logits_kernel(const __bf16* __restrict__ h_last_b,
              const float* __restrict__ W_out,
              const float* __restrict__ b_out,
              float* __restrict__ logits) {
  const int lane = threadIdx.x;
  const int j0 = blockIdx.x * 16;
  const int m0 = blockIdx.y * 16;
  const int jc = j0 + (lane & 15);

  v8f acc = splat8(b_out[jc]);
  const int arow = m0 + (lane & 15);
#pragma unroll 2
  for (int k0 = 0; k0 < H_; k0 += 32) {
    v16bf a = load_a_bf16(h_last_b, arow, H_, k0, lane);
    v16bf b = load_b_f32(W_out, jc, H_, k0, lane);
    acc = wmma_bf16(a, b, acc);
  }
  const int mbase = m0 + ((lane >> 4) << 3);
#pragma unroll
  for (int v = 0; v < 8; ++v)
    logits[(long)(mbase + v) * C_ + jc] = acc[v];
}

// ---------------------------------------------------------------------------
// Cross-entropy (ignore_index=0, mean over kept) + accuracy. 1 block x 256.
// ---------------------------------------------------------------------------
__global__ void __launch_bounds__(256)
loss_kernel(const float* __restrict__ logits,
            const int* __restrict__ labels,
            float* __restrict__ out) {
  __shared__ float s_nll[256];
  __shared__ float s_msk[256];
  __shared__ float s_cor[256];

  const int b = threadIdx.x;
  const float* row = logits + (long)b * C_;

  float mx = row[0];
  int arg = 0;
#pragma unroll 4
  for (int c = 1; c < C_; ++c) {
    float v = row[c];
    if (v > mx) { mx = v; arg = c; }
  }
  float sum = 0.0f;
#pragma unroll 4
  for (int c = 0; c < C_; ++c) sum += __expf(row[c] - mx);
  float lse = mx + __logf(sum);

  int lab = labels[b];
  float mask = (lab != 0) ? 1.0f : 0.0f;
  s_nll[b] = (lse - row[lab]) * mask;
  s_msk[b] = mask;
  s_cor[b] = (arg == lab) ? 1.0f : 0.0f;
  __syncthreads();

  for (int off = 128; off > 0; off >>= 1) {
    if (b < off) {
      s_nll[b] += s_nll[b + off];
      s_msk[b] += s_msk[b + off];
      s_cor[b] += s_cor[b + off];
    }
    __syncthreads();
  }
  if (b == 0) {
    out[0] = s_nll[0] / fmaxf(s_msk[0], 1.0f);
    out[1] = s_cor[0] / (float)B_;
  }
}

// ---------------------------------------------------------------------------
extern "C" void kernel_launch(void* const* d_in, const int* in_sizes, int n_in,
                              void* d_out, int out_size, void* d_ws, size_t ws_size,
                              hipStream_t stream) {
  (void)in_sizes; (void)n_in; (void)out_size;

  const float* data   = (const float*)d_in[0];   // [B,S,E]
  const int*   labels = (const int*)  d_in[1];   // [B]
  const float* h0     = (const float*)d_in[2];   // [B,H]
  const float* W_ih   = (const float*)d_in[3];   // [3H,E]
  const float* W_hh   = (const float*)d_in[4];   // [3H,H]
  const float* b_ih   = (const float*)d_in[5];   // [3H]
  const float* b_hh   = (const float*)d_in[6];   // [3H]
  const float* W_out  = (const float*)d_in[7];   // [C,H]
  const float* b_out  = (const float*)d_in[8];   // [C]

  // Workspace layout. Base (~4.2 MB) always required; Xbf (+64 MB) and GI
  // (+768 MB) enable the faster tiers if ws_size allows.
  char* ws = (char*)d_ws;
  __bf16* Wih_bf = (__bf16*)(ws);                 // 786432 B
  __bf16* Whh_bf = (__bf16*)(ws + 786432);        // 1572864 B  (end 2359296)
  float*  hf0    = (float*) (ws + 2359296);       // 524288 B
  float*  hf1    = (float*) (ws + 2883584);       // 524288 B
  __bf16* hb0    = (__bf16*)(ws + 3407872);       // 262144 B
  __bf16* hb1    = (__bf16*)(ws + 3670016);       // 262144 B
  __bf16* h0_bf  = (__bf16*)(ws + 3932160);       // 262144 B
  float*  logits = (float*) (ws + 4194304);       // 131072 B   (end 4325376)
  const size_t XBF_OFF = 4325376;
  const size_t XBF_SZ  = (size_t)BS_ * E_ * 2;    // 67108864 B
  const size_t GI_OFF  = XBF_OFF + XBF_SZ;        // 71434240
  const size_t GI_SZ   = (size_t)BS_ * G3H_ * 4;  // 805306368 B
  __bf16* Xbf = (__bf16*)(ws + XBF_OFF);
  float*  GI  = (float*) (ws + GI_OFF);

  const bool has_xbf = ws_size >= XBF_OFF + XBF_SZ;
  const bool has_gi  = ws_size >= GI_OFF + GI_SZ;

  convert_inputs<<<(G3H_ * H_ + 255) / 256, 256, 0, stream>>>(
      W_ih, W_hh, h0, Wih_bf, Whh_bf, h0_bf);
  if (has_xbf)
    convert_x<<<(BS_ * E_ + 255) / 256, 256, 0, stream>>>(data, Xbf);

  const dim3 sgrid(H_ / 16, B_ / 32);   // 32 x 8 = 256 waves/step

  if (has_gi) {
    // Tier 1: hoist gi out of the scan entirely.
    gi_gemm<<<dim3(BS_ / 32, G3H_ / 16), 32, 0, stream>>>(Xbf, Wih_bf, b_ih, GI);
    for (int t = 0; t < S_; ++t) {
      const float*  hpf = (t == 0) ? h0    : ((t & 1) ? hf0 : hf1);
      const __bf16* hpb = (t == 0) ? h0_bf : ((t & 1) ? hb0 : hb1);
      float*        hnf = (t & 1) ? hf1 : hf0;
      __bf16*       hnb = (t & 1) ? hb1 : hb0;
      gru_step_gh<<<sgrid, 32, 0, stream>>>(GI, t, hpf, hpb, hnf, hnb,
                                            Whh_bf, b_hh);
    }
  } else {
    // Tier 2/3: fused gi+gh per step.
    for (int t = 0; t < S_; ++t) {
      const float*  hpf = (t == 0) ? h0    : ((t & 1) ? hf0 : hf1);
      const __bf16* hpb = (t == 0) ? h0_bf : ((t & 1) ? hb0 : hb1);
      float*        hnf = (t & 1) ? hf1 : hf0;
      __bf16*       hnb = (t & 1) ? hb1 : hb0;
      if (has_xbf)
        gru_step_fused<true><<<sgrid, 32, 0, stream>>>(
            Xbf + (size_t)t * E_, (long)S_ * E_, hpf, hpb, hnf, hnb,
            Wih_bf, Whh_bf, b_ih, b_hh);
      else
        gru_step_fused<false><<<sgrid, 32, 0, stream>>>(
            data + (size_t)t * E_, (long)S_ * E_, hpf, hpb, hnf, hnb,
            Wih_bf, Whh_bf, b_ih, b_hh);
    }
  }
  const __bf16* h_last_b = hb1;         // t = 511 (odd) wrote buffer 1

  logits_kernel<<<dim3(C_ / 16, B_ / 16), 32, 0, stream>>>(
      h_last_b, W_out, b_out, logits);
  loss_kernel<<<1, 256, 0, stream>>>(logits, labels, (float*)d_out);
}
